// Debias_v2_11862699671616
// MI455X (gfx1250) — matrix operand order repaired and checked by reading, not compiled, not run on hardware
//
#include <hip/hip_runtime.h>
#include <hip/hip_bf16.h>
#include <math.h>
#include <stdint.h>

typedef __attribute__((ext_vector_type(16))) _Float16 v16h;
typedef __attribute__((ext_vector_type(8)))  _Float16 v8h;
typedef __attribute__((ext_vector_type(8)))  float    v8f;
typedef __attribute__((ext_vector_type(4)))  unsigned int v4u;
typedef __attribute__((ext_vector_type(8)))  int      v8i;
typedef __attribute__((ext_vector_type(4)))  int      v4i;

#define OMEGA_C 0.1f
#define SLOPE_C 0.01f
#define SQRT_DIM_M 11.313708498984761f   // sqrt(128)
#define A_STRIDE 264                      // 256 f16 + 8 f16 pad (16B) -> bank-conflict-free

__device__ __forceinline__ float lrelu(float v) { return v > 0.f ? v : SLOPE_C * v; }

__device__ __forceinline__ void gatomic_add(float* p, float v) {
  __hip_atomic_fetch_add(p, v, __ATOMIC_RELAXED, __HIP_MEMORY_SCOPE_AGENT);
}

// ---- WMMA operand loaders (wave32, 16x16x32 f16; ISA 7.12.2 layouts) ----
// A (16x32 f16): lane m=lane&15, half=lane>>4.
//   VGPR 0..3 <- K = k0 + 8*half + {0..7}   (contiguous 16B)
//   VGPR 4..7 <- K = k0 + 16 + 8*half + {0..7}
__device__ __forceinline__ v16h load_a16(const _Float16* row, int k0, int half) {
  v8h lo = *(const v8h*)(row + k0 + 8 * half);
  v8h hi = *(const v8h*)(row + k0 + 16 + 8 * half);
  return __builtin_shufflevector(lo, hi, 0, 1, 2, 3, 4, 5, 6, 7, 8, 9, 10, 11, 12, 13, 14, 15);
}
// B (32x16 f16): lane n=lane&15 reads B[k][n] for K = k0 + 8*half + {0..7}, k0 + 16 + 8*half + {0..7};
// every B matrix is stored so row n is contiguous in K -> two 16B loads.
__device__ __forceinline__ v16h load_b16(const _Float16* __restrict__ wrow, int k0, int half) {
  v8h lo = *(const v8h*)(wrow + k0 + 8 * half);
  v8h hi = *(const v8h*)(wrow + k0 + 16 + 8 * half);
  return __builtin_shufflevector(lo, hi, 0, 1, 2, 3, 4, 5, 6, 7, 8, 9, 10, 11, 12, 13, 14, 15);
}
// A from an f32 row with per-row scale (degree-normalized i_agg).
__device__ __forceinline__ v16h load_a_f32(const float* __restrict__ row, int k0, int half, float scale) {
  const float* p0 = row + k0 + 8 * half;
  const float* p1 = row + k0 + 16 + 8 * half;
  v16h a;
#pragma unroll
  for (int j = 0; j < 8; ++j) a[j] = (_Float16)(p0[j] * scale);
#pragma unroll
  for (int j = 0; j < 8; ++j) a[8 + j] = (_Float16)(p1[j] * scale);
  return a;
}

__device__ __forceinline__ v8f wmma16(v16h a, v16h b, v8f c) {
  return __builtin_amdgcn_wmma_f32_16x16x32_f16(false, a, false, b, (short)0, c, false, false);
}

// ---------------- prep kernels ----------------
__global__ void zero_kernel(float* __restrict__ p, long n) {
  long i = (long)blockIdx.x * blockDim.x + threadIdx.x;
  long stride = (long)gridDim.x * blockDim.x;
  for (; i < n; i += stride) p[i] = 0.f;
}

__global__ void cvt_kernel(const float* __restrict__ in, _Float16* __restrict__ out, long n) {
  long i = (long)blockIdx.x * blockDim.x + threadIdx.x;
  long stride = (long)gridDim.x * blockDim.x;
  for (; i < n; i += stride) out[i] = (_Float16)in[i];
}

// transpose-convert: in is R x C (row-major), out[c*R + r] = in[r*C + c]
__global__ void cvtT_kernel(const float* __restrict__ in, _Float16* __restrict__ out, int R, int C) {
  int i = blockIdx.x * blockDim.x + threadIdx.x;
  if (i >= R * C) return;
  int r = i / C, c = i % C;
  out[(long)c * R + r] = (_Float16)in[(long)r * C + c];
}

__global__ void deg_reduce_kernel(const int* __restrict__ degree, float* __restrict__ deg_sum, int N) {
  __shared__ float red[256];
  int gid = blockIdx.x * 256 + threadIdx.x;
  red[threadIdx.x] = gid < N ? (float)degree[gid] : 0.f;
  __syncthreads();
  for (int s = 128; s > 0; s >>= 1) {
    if (threadIdx.x < s) red[threadIdx.x] += red[threadIdx.x + s];
    __syncthreads();
  }
  if (threadIdx.x == 0) gatomic_add(deg_sum, red[0]);
}

// ---------------- h = (x @ W_w^T + b) * sqrt(DIM_M), fused s_src/s_dst ----------------
// block = 256 thr = 8 waves; block computes 16 rows x 128 cols; wave w -> cols [16w,16w+16)
// A tile (16 x INc f16) staged into LDS by the Tensor Data Mover (padded rows).
__global__ __launch_bounds__(256) void h_gemm_kernel(
    const _Float16* __restrict__ xh, const _Float16* __restrict__ wwh,
    const float* __restrict__ wb, const float* __restrict__ avec,
    float* __restrict__ h, float* __restrict__ s_src, float* __restrict__ s_dst,
    int INc, int OUTc) {
  __shared__ __align__(32) char smem[16 * A_STRIDE * 2 + 2 * 16 * 4];
  _Float16* Atile = (_Float16*)smem;                   // LDS offset 0 (first object)
  float* s0 = (float*)(smem + 16 * A_STRIDE * 2);
  float* s1 = s0 + 16;

  const int tid = threadIdx.x;
  const int lane = tid & 31;
  const int wave = tid >> 5;
  const int m = lane & 15, half = lane >> 4;
  const int n0 = wave * 16;

  if (tid < 16) { s0[tid] = 0.f; s1[tid] = 0.f; }

#if __has_builtin(__builtin_amdgcn_tensor_load_to_lds) && __has_builtin(__builtin_amdgcn_s_wait_tensorcnt)
  if (wave == 0) {
    // Tensor DMA: 1-D tile of 16*INc f16, LDS padding of 4 DW after every 128 DW
    // -> LDS row stride = 528B (A_STRIDE f16), bank-conflict-free A reads.
    unsigned long long ga = (unsigned long long)(uintptr_t)(xh + (size_t)blockIdx.x * 16 * INc);
    unsigned nelem = (unsigned)(16 * INc);             // 4096 f16
    v4u g0;
    g0[0] = 1u;                                        // count=1, user mode, no gather
    g0[1] = 0u;                                        // lds_addr = 0
    g0[2] = (unsigned)(ga & 0xffffffffull);            // global_addr[31:0]
    g0[3] = (unsigned)((ga >> 32) & 0x01ffffffull) | (2u << 30);  // addr[56:32], type=2
    v8i g1;
    g1[0] = (int)((1u << 16) | (1u << 20) | (6u << 22) | (3u << 25));
    //            data=2B     pad_en      128-DW int.   pad 4 DW
    g1[1] = (int)((nelem & 0xffffu) << 16);            // tensor_dim0[15:0]
    g1[2] = (int)((nelem >> 16) | (1u << 16));         // tensor_dim0[31:16], tensor_dim1=1
    g1[3] = (int)((nelem & 0xffffu) << 16);            // tile_dim0 = nelem
    g1[4] = 0;                                         // tile_dim1/2 unused
    g1[5] = (int)nelem;                                // tensor_dim0_stride[31:0]
    g1[6] = 0;
    g1[7] = 0;
    v4i gz = {0, 0, 0, 0};
#if defined(__clang_major__) && (__clang_major__ >= 23)
    v8i gz8 = {0, 0, 0, 0, 0, 0, 0, 0};
    __builtin_amdgcn_tensor_load_to_lds(g0, g1, gz, gz, gz8, 0);
#else
    __builtin_amdgcn_tensor_load_to_lds(g0, g1, gz, gz, 0);
#endif
    __builtin_amdgcn_s_wait_tensorcnt(0);
  }
#else
  {
    // cooperative padded copy fallback (16B chunks)
    const _Float16* src = xh + (size_t)blockIdx.x * 16 * INc;
    int chunks_per_row = INc / 8;
    for (int i = tid; i < 16 * chunks_per_row; i += 256) {
      int r = i / chunks_per_row, c = i % chunks_per_row;
      *(v8h*)(Atile + r * A_STRIDE + c * 8) = *(const v8h*)(src + (long)r * INc + c * 8);
    }
  }
#endif
  __syncthreads();

  const _Float16* arow = Atile + m * A_STRIDE;                 // LDS reads (ds_load_b128)
  const _Float16* brow = wwh + (size_t)(n0 + m) * INc;         // B[k][n] = W_w[n][k]
  v8f c = {};
  for (int k0 = 0; k0 < INc; k0 += 32)
    c = wmma16(load_a16(arow, k0, half), load_b16(brow, k0, half), c);

  float u0, u1;
#pragma unroll
  for (int i = 0; i < 8; ++i) {
    int r = blockIdx.x * 16 + i + 8 * half;    // D VGPR i: M=i (lanes 0-15) / i+8 (16-31)
    int cc = n0 + m;
    float hv = (c[i] + wb[cc]) * SQRT_DIM_M;
    h[(size_t)r * OUTc + cc] = hv;
    // fused s_src/s_dst: reduce hv*a over the 16 lanes of this half
    u0 = hv * avec[cc];
    u1 = hv * avec[OUTc + cc];
#pragma unroll
    for (int off = 8; off >= 1; off >>= 1) {
      u0 += __shfl_xor(u0, off, 16);
      u1 += __shfl_xor(u1, off, 16);
    }
    if (m == 0) {
      atomicAdd(&s0[i + 8 * half], u0);   // ds_add_f32 across waves
      atomicAdd(&s1[i + 8 * half], u1);
    }
  }
  __syncthreads();
  if (tid < 16) {
    s_src[blockIdx.x * 16 + tid] = s0[tid];
    s_dst[blockIdx.x * 16 + tid] = s1[tid];
  }
}

// ---------------- fused edge pass: both segment sums + rowsum ----------------
// one wave per edge; lane handles 4 features (128/32)
__global__ __launch_bounds__(256) void edge_kernel(
    const int* __restrict__ edge, const float* __restrict__ adj,
    const float* __restrict__ h, const float* __restrict__ ssrc, const float* __restrict__ sdst,
    float* __restrict__ i_agg, float* __restrict__ out_acc, float* __restrict__ rowsum, int E) {
  long e = (long)blockIdx.x * 8 + (threadIdx.x >> 5);
  int lane = threadIdx.x & 31;
  if (e >= E) return;
  int src = edge[e], dst = edge[(long)E + e];
  float w0 = adj[e];
  float w1 = __expf(-lrelu(ssrc[src] + sdst[dst]));
  float4 v = ((const float4*)(h + (long)dst * 128))[lane];
  float* ia = i_agg + (long)src * 128 + lane * 4;
  float* oa = out_acc + (long)src * 128 + lane * 4;
  gatomic_add(ia + 0, w0 * v.x); gatomic_add(ia + 1, w0 * v.y);
  gatomic_add(ia + 2, w0 * v.z); gatomic_add(ia + 3, w0 * v.w);
  gatomic_add(oa + 0, w1 * v.x); gatomic_add(oa + 1, w1 * v.y);
  gatomic_add(oa + 2, w1 * v.z); gatomic_add(oa + 3, w1 * v.w);
  if (lane == 0) gatomic_add(rowsum + src, w1);
}

// ---------------- final: gamma/beta/b_add/b_rev tiles + bias + normalize ----------------
__global__ __launch_bounds__(256) void final_kernel(
    const int* __restrict__ degree, const _Float16* __restrict__ peh,
    const float* __restrict__ i_agg,
    const _Float16* __restrict__ wgT, const _Float16* __restrict__ wbT,
    const _Float16* __restrict__ waddh, const _Float16* __restrict__ wrevh,
    const float* __restrict__ b_gamma, const float* __restrict__ b_beta,
    const float* __restrict__ rowsum, const float* __restrict__ deg_sum,
    float* __restrict__ out, int N) {
  const int lane = threadIdx.x & 31;
  const int wave = threadIdx.x >> 5;
  const int m = lane & 15, half = lane >> 4;
  const int row = blockIdx.x * 16 + m;
  const int n0 = wave * 16;
  const int n = n0 + m;
  int dg = degree[row];
  const _Float16* pe_row = peh + (long)dg * 128;
  float inv_deg = dg > 0 ? 1.0f / (float)dg : 0.0f;
  const float* ia_row = i_agg + (long)row * 128;
  v8f cg = {}, cb = {}, ca = {}, cr = {};
#pragma unroll
  for (int k0 = 0; k0 < 128; k0 += 32) {
    v16h ape = load_a16(pe_row, k0, half);
    v16h aia = load_a_f32(ia_row, k0, half, inv_deg);
    cg = wmma16(ape, load_b16(wgT + (long)n * 128, k0, half), cg);
    cb = wmma16(ape, load_b16(wbT + (long)n * 128, k0, half), cb);
    ca = wmma16(aia, load_b16(waddh + (long)n * 128, k0, half), ca);
    cr = wmma16(aia, load_b16(wrevh + (long)n * 128, k0, half), cr);
  }
  float K = deg_sum[0] / (float)N;
#pragma unroll
  for (int i = 0; i < 8; ++i) {
    int r = blockIdx.x * 16 + i + 8 * half;
    int cc = n0 + m;
    float g = lrelu(cg[i] + b_gamma[cc]);
    float be = lrelu(cb[i] + b_beta[cc]);
    float ba = (g + 1.f) * ca[i] + be;
    float br = (g + 1.f) * cr[i] + be;
    float R = ((float)degree[r] < K) ? 1.f : 0.f;
    float bias = OMEGA_C * (R * ba - (1.f - R) * br);
    long o = (long)r * 128 + cc;
    out[o] = (out[o] + bias) / (rowsum[r] + 1e-5f + 1.f);
  }
}

// ---------------- norms over idx rows (L_b, L_film) ----------------
__global__ __launch_bounds__(256) void idx_norm_kernel(
    const int* __restrict__ idx, const int* __restrict__ degree,
    const _Float16* __restrict__ peh, const float* __restrict__ i_agg,
    const _Float16* __restrict__ wgT, const _Float16* __restrict__ wbT,
    const _Float16* __restrict__ waddh, const _Float16* __restrict__ wrevh,
    const float* __restrict__ b_gamma, const float* __restrict__ b_beta,
    const float* __restrict__ deg_sum, float* __restrict__ Lb_acc, float* __restrict__ Lf_acc, int N) {
  __shared__ float sb[16], sg[16], sbe[16];
  const int lane = threadIdx.x & 31;
  const int wave = threadIdx.x >> 5;
  const int m = lane & 15, half = lane >> 4;
  const int n0 = wave * 16;
  const int n = n0 + m;
  if (threadIdx.x < 16) { sb[threadIdx.x] = 0.f; sg[threadIdx.x] = 0.f; sbe[threadIdx.x] = 0.f; }
  __syncthreads();
  int node_m = idx[blockIdx.x * 16 + m];
  int dg = degree[node_m];
  const _Float16* pe_row = peh + (long)dg * 128;
  float inv_deg = dg > 0 ? 1.0f / (float)dg : 0.0f;
  const float* ia_row = i_agg + (long)node_m * 128;
  v8f cg = {}, cb = {}, ca = {}, cr = {};
#pragma unroll
  for (int k0 = 0; k0 < 128; k0 += 32) {
    v16h ape = load_a16(pe_row, k0, half);
    v16h aia = load_a_f32(ia_row, k0, half, inv_deg);
    cg = wmma16(ape, load_b16(wgT + (long)n * 128, k0, half), cg);
    cb = wmma16(ape, load_b16(wbT + (long)n * 128, k0, half), cb);
    ca = wmma16(aia, load_b16(waddh + (long)n * 128, k0, half), ca);
    cr = wmma16(aia, load_b16(wrevh + (long)n * 128, k0, half), cr);
  }
  float K = deg_sum[0] / (float)N;
#pragma unroll
  for (int i = 0; i < 8; ++i) {
    int rl = i + 8 * half;
    int node = idx[blockIdx.x * 16 + rl];
    int cc = n0 + m;
    float g = lrelu(cg[i] + b_gamma[cc]);
    float be = lrelu(cb[i] + b_beta[cc]);
    float ba = (g + 1.f) * ca[i] + be;
    float br = (g + 1.f) * cr[i] + be;
    bool R = (float)degree[node] < K;
    atomicAdd(&sb[rl], R ? ba * ba : br * br);   // ds_add_f32
    atomicAdd(&sg[rl], g * g);
    atomicAdd(&sbe[rl], be * be);
  }
  __syncthreads();
  if (threadIdx.x < 16) {
    gatomic_add(Lb_acc, sqrtf(sb[threadIdx.x]));
    gatomic_add(Lf_acc, sqrtf(sg[threadIdx.x]) + sqrtf(sbe[threadIdx.x]));
  }
}

__global__ void scalars_kernel(const float* __restrict__ Lb, const float* __restrict__ Lf,
                               float* __restrict__ out, long off, float inv_nidx) {
  if (threadIdx.x == 0 && blockIdx.x == 0) {
    out[off] = Lb[0] * inv_nidx;
    out[off + 1] = Lf[0] * inv_nidx;
  }
}

extern "C" void kernel_launch(void* const* d_in, const int* in_sizes, int n_in,
                              void* d_out, int out_size, void* d_ws, size_t ws_size,
                              hipStream_t stream) {
  const int INc = 256, OUTc = 128, DIMM = 128;
  const float* x        = (const float*)d_in[0];
  const int*   edge     = (const int*)d_in[1];
  const float* adj      = (const float*)d_in[2];
  const int*   degree   = (const int*)d_in[3];
  const int*   idx      = (const int*)d_in[4];
  const float* weight_w = (const float*)d_in[5];
  const float* weight_b = (const float*)d_in[6];
  const float* a        = (const float*)d_in[7];
  const float* W_gamma  = (const float*)d_in[8];
  const float* W_beta   = (const float*)d_in[9];
  const float* b_gamma  = (const float*)d_in[10];
  const float* b_beta   = (const float*)d_in[11];
  const float* W_add    = (const float*)d_in[12];
  const float* W_rev    = (const float*)d_in[13];
  const float* PE       = (const float*)d_in[14];

  const int N       = in_sizes[0] / INc;
  const int E       = in_sizes[1] / 2;
  const int NIDX    = in_sizes[4];
  const int PE_ROWS = in_sizes[14] / DIMM;

  // workspace bump allocator (256B aligned)
  char* w = (char*)d_ws;
  auto alloc = [&](size_t bytes) -> char* {
    char* p = w;
    w += (bytes + 255) & ~(size_t)255;
    return p;
  };
  float*     h      = (float*)alloc((size_t)N * OUTc * 4);
  float*     i_agg  = (float*)alloc((size_t)N * OUTc * 4);
  float*     s_src  = (float*)alloc((size_t)N * 4);
  float*     s_dst  = (float*)alloc((size_t)N * 4);
  float*     rowsum = (float*)alloc((size_t)N * 4);
  float*     scal   = (float*)alloc(4 * 4);  // [deg_sum, Lb, Lf, pad]
  _Float16*  xh     = (_Float16*)alloc((size_t)N * INc * 2);
  _Float16*  wwh    = (_Float16*)alloc((size_t)OUTc * INc * 2);
  _Float16*  peh    = (_Float16*)alloc((size_t)PE_ROWS * DIMM * 2);
  _Float16*  wgT    = (_Float16*)alloc((size_t)DIMM * OUTc * 2);
  _Float16*  wbT    = (_Float16*)alloc((size_t)DIMM * OUTc * 2);
  _Float16*  waddh  = (_Float16*)alloc((size_t)OUTc * OUTc * 2);
  _Float16*  wrevh  = (_Float16*)alloc((size_t)OUTc * OUTc * 2);

  float* out = (float*)d_out;

  // zero accumulators + output accumulator
  zero_kernel<<<2048, 256, 0, stream>>>(i_agg, (long)N * OUTc);
  zero_kernel<<<2048, 256, 0, stream>>>(out, (long)N * OUTc);
  zero_kernel<<<256, 256, 0, stream>>>(rowsum, N);
  zero_kernel<<<1, 64, 0, stream>>>(scal, 4);

  // f16 conversions / transposes
  cvt_kernel<<<4096, 256, 0, stream>>>(x, xh, (long)N * INc);
  cvt_kernel<<<64, 256, 0, stream>>>(weight_w, wwh, (long)OUTc * INc);
  cvt_kernel<<<128, 256, 0, stream>>>(PE, peh, (long)PE_ROWS * DIMM);
  cvt_kernel<<<64, 256, 0, stream>>>(W_add, waddh, (long)OUTc * OUTc);
  cvt_kernel<<<64, 256, 0, stream>>>(W_rev, wrevh, (long)OUTc * OUTc);
  cvtT_kernel<<<(DIMM * OUTc + 255) / 256, 256, 0, stream>>>(W_gamma, wgT, DIMM, OUTc);
  cvtT_kernel<<<(DIMM * OUTc + 255) / 256, 256, 0, stream>>>(W_beta, wbT, DIMM, OUTc);

  deg_reduce_kernel<<<(N + 255) / 256, 256, 0, stream>>>(degree, scal, N);

  h_gemm_kernel<<<N / 16, 256, 0, stream>>>(xh, wwh, weight_b, a, h, s_src, s_dst, INc, OUTc);
  edge_kernel<<<(E + 7) / 8, 256, 0, stream>>>(edge, adj, h, s_src, s_dst, i_agg, out, rowsum, E);
  final_kernel<<<N / 16, 256, 0, stream>>>(degree, peh, i_agg, wgT, wbT, waddh, wrevh,
                                           b_gamma, b_beta, rowsum, scal, out, N);
  idx_norm_kernel<<<NIDX / 16, 256, 0, stream>>>(idx, degree, peh, i_agg, wgT, wbT, waddh, wrevh,
                                                 b_gamma, b_beta, scal, scal + 1, scal + 2, N);
  scalars_kernel<<<1, 32, 0, stream>>>(scal + 1, scal + 2, out, (long)N * OUTc, 1.0f / (float)NIDX);
}
